// xLSTMLayer_61332132987573
// MI455X (gfx1250) — compile-verified, matrix operand order checked by reading, not run on hardware
//
#include <hip/hip_runtime.h>
#include <hip/hip_bf16.h>

// ---------------- CDNA5 WMMA types ----------------
typedef __attribute__((ext_vector_type(16))) __bf16 bf16x16;
typedef __attribute__((ext_vector_type(8)))  float  f32x8;

#define L_SEQ   2048
#define D_MODEL 1024
#define D_INNER 2048
#define N_HEADS 8
#define DH      256
#define D_PROJ  4096   // 2*D_INNER
#define D_HID   2048

// RNE pack two f32 -> one dword of 2 bf16 (lo = first)
__device__ __forceinline__ unsigned pack2_bf16(float a, float b) {
  unsigned ua = __float_as_uint(a);
  unsigned ub = __float_as_uint(b);
  ua += 0x7FFFu + ((ua >> 16) & 1u);
  ub += 0x7FFFu + ((ub >> 16) & 1u);
  return ((ua >> 16) & 0xFFFFu) | (ub & 0xFFFF0000u);
}

union FragAB { unsigned u[8]; bf16x16 v; };

// ---------------------------------------------------------------------------
// Generic batched GEMM:  C = alpha * A(MxK) @ B + bias + add
//   A: f32 row-major, leading dim lda, per-batch stride sA
//   B: if bT==1: stored (N,K) row-major (weight.T style); else (K,N) row-major
//   C: f32 row-major ldc. Requires M%128==0, N%128==0, K%32==0 (all call sites
//   satisfy this). fp32 staged to bf16 in LDS; v_wmma_f32_16x16x32_bf16.
// ---------------------------------------------------------------------------
__global__ __launch_bounds__(256) void gemm_bf16_wmma(
    int M, int N, int K,
    const float* __restrict__ A, int lda, long long sA,
    const float* __restrict__ B, int ldb, long long sB, int bT,
    float* __restrict__ C, int ldc, long long sC,
    const float* __restrict__ bias,
    const float* __restrict__ add, int ldadd,
    float alpha)
{
  __shared__ unsigned As[128][17];   // [m][kpair], padded stride vs 64 banks
  __shared__ unsigned Bs[128][17];   // [n][kpair]

  const int t    = threadIdx.x;
  const int lane = t & 31;
  const int wid  = t >> 5;
  const int wm   = (wid >> 2) * 64;  // wave row offset inside tile (2 waves in M)
  const int wn   = (wid & 3) * 32;   // wave col offset inside tile (4 waves in N)
  const int m0   = blockIdx.y * 128;
  const int n0   = blockIdx.x * 128;
  const int z    = blockIdx.z;

  A += (long long)z * sA;
  B += (long long)z * sB;
  C += (long long)z * sC;

  f32x8 acc[4][2] = {};

  for (int k0 = 0; k0 < K; k0 += 32) {
    __syncthreads();
    // ---- stage A tile 128x32 f32 -> bf16 (8 dwords per thread) ----
    #pragma unroll
    for (int i = 0; i < 8; ++i) {
      int u  = t * 8 + i;          // 0..2047
      int r  = u >> 4;             // row 0..127
      int kp = u & 15;             // k-pair 0..15
      const float* s = A + (long long)(m0 + r) * lda + (k0 + kp * 2);
      As[r][kp] = pack2_bf16(s[0], s[1]);
    }
    // ---- stage B tile (as [n][k]) ----
    #pragma unroll
    for (int i = 0; i < 8; ++i) {
      int u  = t * 8 + i;
      int r  = u >> 4;             // n 0..127
      int kp = u & 15;
      float b0, b1;
      if (bT) {
        const float* s = B + (long long)(n0 + r) * ldb + (k0 + kp * 2);
        b0 = s[0]; b1 = s[1];
      } else {
        const float* s = B + (long long)(k0 + kp * 2) * ldb + (n0 + r);
        b0 = s[0]; b1 = s[ldb];
      }
      Bs[r][kp] = pack2_bf16(b0, b1);
    }
    // speculative prefetch of next K tile (global_prefetch_b8)
    if (k0 + 32 < K) {
      __builtin_prefetch(A + (long long)(m0 + (t >> 1)) * lda + (k0 + 32), 0, 1);
    }
    __syncthreads();

    // ---- fragments per the 16-bit WMMA VGPR layout ----
    FragAB af[4], bfg[2];
    #pragma unroll
    for (int ti = 0; ti < 4; ++ti) {
      int mr = wm + ti * 16 + (lane & 15);
      #pragma unroll
      for (int d = 0; d < 8; ++d) {
        int kp = ((d >> 2) << 3) + ((lane >> 4) << 2) + (d & 3);
        af[ti].u[d] = As[mr][kp];
      }
    }
    #pragma unroll
    for (int tj = 0; tj < 2; ++tj) {
      int nr = wn + tj * 16 + (lane & 15);
      #pragma unroll
      for (int d = 0; d < 8; ++d) {
        int kp = ((d >> 2) << 3) + ((lane >> 4) << 2) + (d & 3);
        bfg[tj].u[d] = Bs[nr][kp];
      }
    }
    #pragma unroll
    for (int ti = 0; ti < 4; ++ti)
      #pragma unroll
      for (int tj = 0; tj < 2; ++tj)
        acc[ti][tj] = __builtin_amdgcn_wmma_f32_16x16x32_bf16(
            false, af[ti].v, false, bfg[tj].v, (short)0, acc[ti][tj],
            false, false);
  }

  // ---- epilogue: C/D layout: dword r -> row r + (lane>=16)*8, col lane&15 ----
  #pragma unroll
  for (int ti = 0; ti < 4; ++ti) {
    #pragma unroll
    for (int tj = 0; tj < 2; ++tj) {
      #pragma unroll
      for (int r = 0; r < 8; ++r) {
        int row = m0 + wm + ti * 16 + r + ((lane >> 4) << 3);
        int col = n0 + wn + tj * 16 + (lane & 15);
        float v = acc[ti][tj][r] * alpha;
        if (bias) v += bias[col];
        if (add)  v += add[(long long)row * ldadd + col];
        C[(long long)row * ldc + col] = v;
      }
    }
  }
}

// ---------------- LayerNorm (weight only, eps 1e-5) ----------------
__global__ __launch_bounds__(256) void ln_kernel(const float* __restrict__ x,
                                                 const float* __restrict__ w,
                                                 float* __restrict__ out, int D)
{
  int row = blockIdx.x, t = threadIdx.x;
  const float* xr = x + (long long)row * D;
  float s = 0.f, s2 = 0.f;
  for (int j = t; j < D; j += 256) { float v = xr[j]; s += v; s2 += v * v; }
  __shared__ float sh[256], sh2[256];
  sh[t] = s; sh2[t] = s2; __syncthreads();
  for (int o = 128; o > 0; o >>= 1) {
    if (t < o) { sh[t] += sh[t + o]; sh2[t] += sh2[t + o]; }
    __syncthreads();
  }
  float mean = sh[0] / D;
  float var  = sh2[0] / D - mean * mean;
  float inv  = rsqrtf(var + 1e-5f);
  for (int j = t; j < D; j += 256)
    out[(long long)row * D + j] = (xr[j] - mean) * inv * w[j];
}

// ---------------- causal depthwise conv (D_CONV=4) + SiLU ----------------
__global__ __launch_bounds__(256) void conv_silu_kernel(
    const float* __restrict__ xi, int ldxi,
    const float* __restrict__ ck, const float* __restrict__ cb,
    float* __restrict__ xc)
{
  long long idx = (long long)blockIdx.x * 256 + threadIdx.x;  // L*D_INNER
  int c = (int)(idx & (D_INNER - 1));
  int l = (int)(idx >> 11);
  float acc = cb[c];
  #pragma unroll
  for (int j = 0; j < 4; ++j) {
    int src = l + j - 3;
    if (src >= 0) acc += xi[(long long)src * ldxi + c] * ck[c * 4 + (3 - j)];
  }
  xc[idx] = acc / (1.f + __expf(-acc)) ;  // silu
}

// ---------------- gates: ig/fg[h,l] from concat(q,k,v) ----------------
__global__ __launch_bounds__(256) void gate_kernel(
    const float* __restrict__ q, const float* __restrict__ k,
    const float* __restrict__ v,
    const float* __restrict__ igw, const float* __restrict__ igb,
    const float* __restrict__ fgw, const float* __restrict__ fgb,
    float* __restrict__ ig, float* __restrict__ fg)
{
  int l = blockIdx.x, t = threadIdx.x;
  float ia[N_HEADS] = {}, fa[N_HEADS] = {};
  for (int j = t; j < D_INNER; j += 256) {
    float qv = q[(long long)l * D_INNER + j];
    float kv = k[(long long)l * D_INNER + j];
    float vv = v[(long long)l * D_INNER + j];
    #pragma unroll
    for (int h = 0; h < N_HEADS; ++h) {
      const float* iw = igw + h * (3 * D_INNER);
      const float* fw = fgw + h * (3 * D_INNER);
      ia[h] += qv * iw[j] + kv * iw[D_INNER + j] + vv * iw[2 * D_INNER + j];
      fa[h] += qv * fw[j] + kv * fw[D_INNER + j] + vv * fw[2 * D_INNER + j];
    }
  }
  __shared__ float sh[256];
  for (int h = 0; h < N_HEADS; ++h) {
    sh[t] = ia[h]; __syncthreads();
    for (int o = 128; o > 0; o >>= 1) { if (t < o) sh[t] += sh[t + o]; __syncthreads(); }
    if (t == 0) ig[h * L_SEQ + l] = sh[0] + igb[h];
    __syncthreads();
    sh[t] = fa[h]; __syncthreads();
    for (int o = 128; o > 0; o >>= 1) { if (t < o) sh[t] += sh[t + o]; __syncthreads(); }
    if (t == 0) fg[h * L_SEQ + l] = sh[0] + fgb[h];
    __syncthreads();
  }
}

// ---- per-head scans: cs = cumsum(logsigmoid(fg)); g[m]=ig[m]-cs[m+1];
//      pm = prefix max of g; en[l] = exp(-(cs[l+1]+pm[l])) = exp(-max_log_D)
__global__ void mlstm_prep_kernel(const float* __restrict__ ig,
                                  const float* __restrict__ fg,
                                  float* __restrict__ g,
                                  float* __restrict__ pm,
                                  float* __restrict__ en)
{
  int h = blockIdx.x;
  if (threadIdx.x != 0) return;
  float cs = 0.f, run = -__builtin_inff();
  for (int m = 0; m < L_SEQ; ++m) {
    float f  = fg[h * L_SEQ + m];
    float ls = fminf(f, 0.f) - log1pf(__expf(-fabsf(f)));  // log_sigmoid
    cs += ls;                                              // cs[m+1]
    float gm = ig[h * L_SEQ + m] - cs;
    g[h * L_SEQ + m]  = gm;
    run = fmaxf(run, gm);
    pm[h * L_SEQ + m] = run;
    en[h * L_SEQ + m] = __expf(-(cs + run));
  }
}

// ---- row-wise: c = scores*D, norm, write back c/(norm+1e-6) in place ----
__global__ __launch_bounds__(256) void score_post_kernel(
    float* __restrict__ sc, const float* __restrict__ g,
    const float* __restrict__ pm, const float* __restrict__ en)
{
  int l = blockIdx.x, t = threadIdx.x;
  float pml = pm[l];
  float cv[8]; float s = 0.f;
  #pragma unroll
  for (int i = 0; i < 8; ++i) {
    int m = t + i * 256;
    float c = 0.f;
    if (m <= l) c = sc[(long long)l * L_SEQ + m] * __expf(g[m] - pml);
    cv[i] = c; s += c;
  }
  __shared__ float sh[256];
  sh[t] = s; __syncthreads();
  for (int o = 128; o > 0; o >>= 1) { if (t < o) sh[t] += sh[t + o]; __syncthreads(); }
  float norm  = fmaxf(fabsf(sh[0]), en[l]);
  float scale = 1.f / (norm + 1e-6f);
  #pragma unroll
  for (int i = 0; i < 8; ++i) {
    int m = t + i * 256;
    sc[(long long)l * L_SEQ + m] = cv[i] * scale;
  }
}

// ---- GroupNorm(8 groups) + skip*xc, fused with * silu(res) ----
__global__ __launch_bounds__(256) void gn_fuse_kernel(
    const float* __restrict__ hh, const float* __restrict__ gw,
    const float* __restrict__ gb, const float* __restrict__ skip,
    const float* __restrict__ xc, const float* __restrict__ proj,
    float* __restrict__ y)
{
  int l = blockIdx.x, t = threadIdx.x;
  int lane = t & 31, w = t >> 5;                 // wave w <-> head group w
  const float* hr = hh + (long long)l * D_INNER + w * DH;
  float s = 0.f, s2 = 0.f;
  for (int j = lane; j < DH; j += 32) { float v = hr[j]; s += v; s2 += v * v; }
  for (int o = 16; o > 0; o >>= 1) { s += __shfl_xor(s, o); s2 += __shfl_xor(s2, o); }
  float mean = s / (float)DH;
  float var  = s2 / (float)DH - mean * mean;
  float inv  = rsqrtf(var + 1e-5f);
  for (int j = lane; j < DH; j += 32) {
    int col  = w * DH + j;
    float hn = (hr[j] - mean) * inv;
    float h2 = hn * gw[col] + gb[col] + skip[col] * xc[(long long)l * D_INNER + col];
    float r  = proj[(long long)l * D_PROJ + D_INNER + col];   // res half
    float sl = r / (1.f + __expf(-r));                        // silu(res)
    y[(long long)l * D_INNER + col] = h2 * sl;
  }
}

// ---- GEGLU: t = gelu_tanh(p[:, :2048]) * p[:, 2048:] ----
__global__ __launch_bounds__(256) void geglu_kernel(const float* __restrict__ p,
                                                    float* __restrict__ tout)
{
  long long idx = (long long)blockIdx.x * 256 + threadIdx.x;  // L*D_HID
  int col = (int)(idx & (D_HID - 1));
  long long l = idx >> 11;
  float a = p[l * D_PROJ + col];
  float z = p[l * D_PROJ + D_HID + col];
  float g = 0.5f * a * (1.f + tanhf(0.7978845608f * (a + 0.044715f * a * a * a)));
  tout[idx] = g * z;
}

// ---------------------------------------------------------------------------
extern "C" void kernel_launch(void* const* d_in, const int* in_sizes, int n_in,
                              void* d_out, int out_size, void* d_ws, size_t ws_size,
                              hipStream_t stream)
{
  const float* x       = (const float*)d_in[0];
  // d_in[1] = mask (causal, implicit)
  const float* ln1_w   = (const float*)d_in[2];
  const float* win_w   = (const float*)d_in[3];
  const float* win_b   = (const float*)d_in[4];
  const float* wout_w  = (const float*)d_in[5];
  const float* conv_k  = (const float*)d_in[6];
  const float* conv_b  = (const float*)d_in[7];
  const float* wq_w    = (const float*)d_in[8];
  const float* wk_w    = (const float*)d_in[9];
  const float* wv_w    = (const float*)d_in[10];
  const float* igate_w = (const float*)d_in[11];
  const float* igate_b = (const float*)d_in[12];
  const float* fgate_w = (const float*)d_in[13];
  const float* fgate_b = (const float*)d_in[14];
  const float* gn_w    = (const float*)d_in[15];
  const float* gn_b    = (const float*)d_in[16];
  const float* skip    = (const float*)d_in[17];
  const float* ln2_w   = (const float*)d_in[18];
  const float* ffn_win = (const float*)d_in[19];
  const float* ffn_wout= (const float*)d_in[20];
  float* out = (float*)d_out;

  // workspace layout (floats), ~151 MB total; buffers re-used across phases
  float* W    = (float*)d_ws;
  float* hbuf = W;                                 // 2048*1024 (LN1 out; later LN2 out)
  float* proj = hbuf + (long long)L_SEQ * D_MODEL; // 2048*4096 (proj; later FFN p)
  float* xc   = proj + (long long)L_SEQ * D_PROJ;  // 2048*2048
  float* qb   = xc   + (long long)L_SEQ * D_INNER; // q; later y = h2*silu(res)
  float* kb   = qb   + (long long)L_SEQ * D_INNER;
  float* vb   = kb   + (long long)L_SEQ * D_INNER; // v; later GEGLU out
  float* scb  = vb   + (long long)L_SEQ * D_INNER; // per-head scores (reused)
  float* hhb  = scb  + (long long)L_SEQ * L_SEQ;   // attention out
  float* xmid = hhb  + (long long)L_SEQ * D_INNER; // 2048*1024
  float* igv  = xmid + (long long)L_SEQ * D_MODEL; // 8*2048 each below
  float* fgv  = igv + N_HEADS * L_SEQ;
  float* gbuf = fgv + N_HEADS * L_SEQ;
  float* pmb  = gbuf + N_HEADS * L_SEQ;
  float* enb  = pmb + N_HEADS * L_SEQ;

  const dim3 blk(256);

  // 1. LN1
  ln_kernel<<<dim3(L_SEQ), blk, 0, stream>>>(x, ln1_w, hbuf, D_MODEL);

  // 2. proj = h @ win_w.T + win_b   (2048x1024 @ 1024x4096)
  gemm_bf16_wmma<<<dim3(D_PROJ/128, L_SEQ/128, 1), blk, 0, stream>>>(
      L_SEQ, D_PROJ, D_MODEL, hbuf, D_MODEL, 0, win_w, D_MODEL, 0, 1,
      proj, D_PROJ, 0, win_b, nullptr, 0, 1.0f);

  // 3. causal conv + SiLU on xi = proj[:, :2048]
  conv_silu_kernel<<<dim3((L_SEQ*D_INNER)/256), blk, 0, stream>>>(
      proj, D_PROJ, conv_k, conv_b, xc);

  // 4-6. per-head q,k (from xc), v (from xi) — batched over heads via grid.z
  gemm_bf16_wmma<<<dim3(DH/128, L_SEQ/128, N_HEADS), blk, 0, stream>>>(
      L_SEQ, DH, DH, xc, D_INNER, DH, wq_w, DH, (long long)DH*DH, 0,
      qb, D_INNER, DH, nullptr, nullptr, 0, 1.0f);
  gemm_bf16_wmma<<<dim3(DH/128, L_SEQ/128, N_HEADS), blk, 0, stream>>>(
      L_SEQ, DH, DH, xc, D_INNER, DH, wk_w, DH, (long long)DH*DH, 0,
      kb, D_INNER, DH, nullptr, nullptr, 0, 1.0f);
  gemm_bf16_wmma<<<dim3(DH/128, L_SEQ/128, N_HEADS), blk, 0, stream>>>(
      L_SEQ, DH, DH, proj, D_PROJ, DH, wv_w, DH, (long long)DH*DH, 0,
      vb, D_INNER, DH, nullptr, nullptr, 0, 1.0f);

  // 7. gates
  gate_kernel<<<dim3(L_SEQ), blk, 0, stream>>>(
      qb, kb, vb, igate_w, igate_b, fgate_w, fgate_b, igv, fgv);

  // 8. per-head scans (g, prefix max, exp(-max_log_D))
  mlstm_prep_kernel<<<dim3(N_HEADS), dim3(32), 0, stream>>>(igv, fgv, gbuf, pmb, enb);

  // 9. per-head attention: scores -> rescale -> @v (score buffer reused)
  for (int h = 0; h < N_HEADS; ++h) {
    gemm_bf16_wmma<<<dim3(L_SEQ/128, L_SEQ/128, 1), blk, 0, stream>>>(
        L_SEQ, L_SEQ, DH, qb + h*DH, D_INNER, 0, kb + h*DH, D_INNER, 0, 1,
        scb, L_SEQ, 0, nullptr, nullptr, 0, 0.0625f /* 1/sqrt(256) */);
    score_post_kernel<<<dim3(L_SEQ), blk, 0, stream>>>(
        scb, gbuf + h*L_SEQ, pmb + h*L_SEQ, enb + h*L_SEQ);
    gemm_bf16_wmma<<<dim3(DH/128, L_SEQ/128, 1), blk, 0, stream>>>(
        L_SEQ, DH, L_SEQ, scb, L_SEQ, 0, vb + h*DH, D_INNER, 0, 0,
        hhb + h*DH, D_INNER, 0, nullptr, nullptr, 0, 1.0f);
  }

  // 10. GroupNorm + skip*xc, * silu(res) -> y (reuses qb)
  gn_fuse_kernel<<<dim3(L_SEQ), blk, 0, stream>>>(
      hhb, gn_w, gn_b, skip, xc, proj, qb);

  // 11. x_mid = x + y @ wout_w.T
  gemm_bf16_wmma<<<dim3(D_MODEL/128, L_SEQ/128, 1), blk, 0, stream>>>(
      L_SEQ, D_MODEL, D_INNER, qb, D_INNER, 0, wout_w, D_INNER, 0, 1,
      xmid, D_MODEL, 0, nullptr, x, D_MODEL, 1.0f);

  // 12. LN2 (reuses hbuf)
  ln_kernel<<<dim3(L_SEQ), blk, 0, stream>>>(xmid, ln2_w, hbuf, D_MODEL);

  // 13. p = hf @ ffn_win.T (reuses proj)
  gemm_bf16_wmma<<<dim3(D_PROJ/128, L_SEQ/128, 1), blk, 0, stream>>>(
      L_SEQ, D_PROJ, D_MODEL, hbuf, D_MODEL, 0, ffn_win, D_MODEL, 0, 1,
      proj, D_PROJ, 0, nullptr, nullptr, 0, 1.0f);

  // 14. GEGLU (reuses vb)
  geglu_kernel<<<dim3((L_SEQ*D_HID)/256), blk, 0, stream>>>(proj, vb);

  // 15. out = x_mid + t @ ffn_wout.T
  gemm_bf16_wmma<<<dim3(D_MODEL/128, L_SEQ/128, 1), blk, 0, stream>>>(
      L_SEQ, D_MODEL, D_HID, vb, D_HID, 0, ffn_wout, D_HID, 0, 1,
      out, D_MODEL, 0, nullptr, xmid, D_MODEL, 1.0f);
}
